// GAT_Model_74337293959432
// MI455X (gfx1250) — matrix operand order repaired and checked by reading, not compiled, not run on hardware
//
#include <hip/hip_runtime.h>

// ---------------- problem constants (fixed by the reference) ----------------
#define GN 20000   // nodes
#define GF 128     // input features
#define GH 128     // hidden
#define GL 4       // layers
#define GO 64      // output
#define GE 640000  // edges (before self loops)
#define NEG_SLOPE 0.2f
#define MT 8       // M-tiles per GEMM block (B held in registers across them)

typedef __attribute__((ext_vector_type(16))) _Float16 v16h;
typedef __attribute__((ext_vector_type(8)))  _Float16 v8h;
typedef __attribute__((ext_vector_type(4)))  _Float16 v4h;
typedef __attribute__((ext_vector_type(8)))  float    v8f;
typedef __attribute__((ext_vector_type(4)))  float    v4f;
typedef __attribute__((ext_vector_type(4)))  unsigned v4u;

// ---------------- helpers ----------------
__device__ __forceinline__ unsigned fkey(float f) {
    unsigned u = __float_as_uint(f);
    return (u & 0x80000000u) ? ~u : (u | 0x80000000u);
}
__device__ __forceinline__ float keyf(unsigned u) {
    return (u & 0x80000000u) ? __uint_as_float(u & 0x7fffffffu)
                             : __uint_as_float(~u);
}

// ---------------- vectorized utility kernels (counts divisible by 4) --------
__global__ void fill4_f32(float* p, float v, int n4) {
    int i = blockIdx.x * blockDim.x + threadIdx.x;
    if (i < n4) { v4f q = {v, v, v, v}; *(v4f*)(p + 4 * (size_t)i) = q; }
}
__global__ void fill4_u32(unsigned* p, unsigned v, int n4) {
    int i = blockIdx.x * blockDim.x + threadIdx.x;
    if (i < n4) { v4u q = {v, v, v, v}; *(v4u*)(p + 4 * (size_t)i) = q; }
}
__global__ void cvt4_f16(_Float16* dst, const float* src, int n4) {
    int i = blockIdx.x * blockDim.x + threadIdx.x;
    if (i >= n4) return;
    v4f s = *(const v4f*)(src + 4 * (size_t)i);
    v4h d;
#pragma unroll
    for (int k = 0; k < 4; ++k) d[k] = (_Float16)s[k];
    *(v4h*)(dst + 4 * (size_t)i) = d;
}

// ---- repack W [128, Nc] f32 (row-major) into WMMA B-fragment layout -------
// dst[((ntile*4 + ktile)*32 + lane)*16 + e] = W[(ktile*32 + 16*(lane>>4) + e)*Nc
//                                               + ntile*16 + (lane&15)]
__global__ void repack_w(_Float16* __restrict__ dst, const float* __restrict__ W,
                         int Nc) {
    int i = blockIdx.x * blockDim.x + threadIdx.x;
    if (i >= 128 * Nc) return;
    int e    = i & 15;
    int lane = (i >> 4) & 31;
    int frag = i >> 9;
    int kt   = frag & 3;
    int nt   = frag >> 2;
    int col  = nt * 16 + (lane & 15);
    int k    = kt * 32 + 16 * (lane >> 4) + e;
    dst[i] = (_Float16)W[(size_t)k * Nc + col];
}

// ---------------- WMMA GEMM: C[M,Nc] = A[M,128] * Bswz (+bias) -------------
// A f16 row-major (K=128); Bswz pre-swizzled fragments; C f32 row-major.
// Block = 128 threads = 4 waves; each wave owns a 16-column strip and MT
// 16-row tiles; B fragments live in registers across all MT tiles.
// Nc is a compile-time constant so all strides become immediate offsets.
// grid = (ceil(Mtiles/MT), Nc/64).
template <int Nc, bool HasBias>
__global__ void gemm_wmma_swz(const _Float16* __restrict__ A,
                              const _Float16* __restrict__ Bswz,
                              float* __restrict__ C,
                              const float* __restrict__ bias,
                              int Mtiles) {
    const int K    = 128;
    const int lane = threadIdx.x & 31;
    const int wave = threadIdx.x >> 5;
    const int n0   = blockIdx.y * 64 + wave * 16;
    const int col  = n0 + (lane & 15);
    const int hiA  = (lane >= 16) ? 16 : 0;  // A K-offset for upper half-wave
    const int rhi  = (lane >> 4) * 8;        // C row offset for upper half-wave

    // ---- load the wave's 4 B fragments (contiguous 32B per lane each) ----
    v16h bf[4];
#pragma unroll
    for (int kt = 0; kt < 4; ++kt) {
        const _Float16* bp =
            Bswz + ((size_t)(((n0 >> 4) * 4 + kt) * 32 + lane)) * 16;
        bf[kt] = *(const v16h*)bp;
    }

    float badd = 0.0f;
    if constexpr (HasBias) badd = bias[col];

#pragma unroll
    for (int t = 0; t < MT; ++t) {
        const int mtile = blockIdx.x * MT + t;
        if (mtile >= Mtiles) continue;   // uniform per-wave predicate
        const int m0  = mtile * 16;
        const int row = m0 + (lane & 15);
        const _Float16* Arow = A + (size_t)row * K + hiA;

        // prefetch next tile's A rows (emits global_prefetch_b8)
        if (t + 1 < MT && mtile + 1 < Mtiles)
            __builtin_prefetch(Arow + (size_t)16 * K, 0, 1);

        // ---- load all 4 A fragments into distinct registers first so the
        //      WMMA chain can overlap with outstanding loads ----
        v16h a[4];
#pragma unroll
        for (int kt = 0; kt < 4; ++kt) {
            const _Float16* Ap = Arow + kt * 32;
            v8h lo = *(const v8h*)(Ap);        // halves [kk .. kk+7]
            v8h hi = *(const v8h*)(Ap + 8);    // halves [kk+8 .. kk+15]
#pragma unroll
            for (int e = 0; e < 8; ++e) { a[kt][e] = lo[e]; a[kt][8 + e] = hi[e]; }
        }

        v8f acc = {};
#pragma unroll
        for (int kt = 0; kt < 4; ++kt)
            acc = __builtin_amdgcn_wmma_f32_16x16x32_f16(
                false, a[kt], false, bf[kt], (short)0, acc, false, false);

        // ---- store: single base + compile-time immediate row offsets ----
        float* Cp = C + (size_t)(m0 + rhi) * Nc + col;
        if constexpr (HasBias) {
#pragma unroll
            for (int v = 0; v < 8; ++v)
                Cp[(size_t)v * Nc] = acc[v] + badd;
        } else {
#pragma unroll
            for (int v = 0; v < 8; ++v)
                Cp[(size_t)v * Nc] = acc[v];
        }
    }
    (void)badd;
}

// ---------------- alpha_s / alpha_d: one wave per node ----------------------
__global__ void alphas_kernel(const float* __restrict__ z,
                              const float* __restrict__ a_s,
                              const float* __restrict__ a_d,
                              float* __restrict__ as_out,
                              float* __restrict__ ad_out) {
    int w = (blockIdx.x * blockDim.x + threadIdx.x) >> 5;
    int lane = threadIdx.x & 31;
    if (w >= GN) return;
    const float* zr = z + (size_t)w * GH;
    float ds = 0.f, dd = 0.f;
#pragma unroll
    for (int j = lane; j < GH; j += 32) {
        float v = zr[j];
        ds += v * a_s[j];
        dd += v * a_d[j];
    }
#pragma unroll
    for (int off = 16; off > 0; off >>= 1) {
        ds += __shfl_xor(ds, off, 32);
        dd += __shfl_xor(dd, off, 32);
    }
    if (lane == 0) { as_out[w] = ds; ad_out[w] = dd; }
}

// ---------------- pass 1: logits + segment max (uint-key atomicMax) ---------
__global__ void edge_logits_kernel(const int* __restrict__ src,
                                   const int* __restrict__ dst,
                                   const float* __restrict__ as_in,
                                   const float* __restrict__ ad_in,
                                   float* __restrict__ logits,
                                   unsigned* __restrict__ mkey) {
    int i = blockIdx.x * blockDim.x + threadIdx.x;
    const int total = GE + GN;
    if (i >= total) return;
    int s, d;
    if (i < GE) { s = src[i]; d = dst[i]; } else { s = d = i - GE; }
    float x = as_in[s] + ad_in[d];
    float lg = (x > 0.f) ? x : NEG_SLOPE * x;
    logits[i] = lg;
    atomicMax(&mkey[d], fkey(lg));
}

// ---------------- pass 2: p=exp(l-m); s[dst]+=p; accum[dst]+=p*z[src] -------
// one wave per edge; lane L covers feature columns 4L..4L+3 (one b128 gather)
__global__ void edge_accum_kernel(const int* __restrict__ src,
                                  const int* __restrict__ dst,
                                  const float* __restrict__ logits,
                                  const unsigned* __restrict__ mkey,
                                  const float* __restrict__ z,
                                  float* __restrict__ ssum,
                                  float* __restrict__ accum) {
    int w = (blockIdx.x * blockDim.x + threadIdx.x) >> 5;
    int lane = threadIdx.x & 31;
    const int total = GE + GN;
    if (w >= total) return;
    int s, d;
    if (w < GE) { s = src[w]; d = dst[w]; } else { s = d = w - GE; }
    float p = __expf(logits[w] - keyf(mkey[d]));
    if (lane == 0) atomicAdd(&ssum[d], p);
    const int j0 = lane * 4;
    v4f zv = *(const v4f*)(z + (size_t)s * GH + j0);
    float* ap = accum + (size_t)d * GH + j0;
    atomicAdd(ap + 0, p * zv[0]);
    atomicAdd(ap + 1, p * zv[1]);
    atomicAdd(ap + 2, p * zv[2]);
    atomicAdd(ap + 3, p * zv[3]);
}

// ------- pass 3: h = relu(accum/(s+eps) + b); also emit f16 copy ------------
// 4 elements per thread (b128 load / b128 + b64 stores)
__global__ void finalize_kernel(const float* __restrict__ accum,
                                const float* __restrict__ ssum,
                                const float* __restrict__ bias,
                                float* __restrict__ hout,
                                _Float16* __restrict__ hout16) {
    int i = blockIdx.x * blockDim.x + threadIdx.x;
    if (i >= GN * GH / 4) return;
    int base = i * 4;
    int n = base >> 7;        // /128
    int j = base & 127;
    float inv = 1.0f / (ssum[n] + 1e-16f);
    v4f av = *(const v4f*)(accum + base);
    v4f bv = *(const v4f*)(bias + j);
    v4f hv;
    v4h h16v;
#pragma unroll
    for (int k = 0; k < 4; ++k) {
        float v = av[k] * inv + bv[k];
        v = v > 0.f ? v : 0.f;
        hv[k] = v;
        h16v[k] = (_Float16)v;
    }
    *(v4f*)(hout + base) = hv;
    *(v4h*)(hout16 + base) = h16v;
}

// ---------------- host-side orchestration ----------------------------------
extern "C" void kernel_launch(void* const* d_in, const int* in_sizes, int n_in,
                              void* d_out, int out_size, void* d_ws, size_t ws_size,
                              hipStream_t stream) {
    (void)in_sizes; (void)n_in; (void)out_size; (void)ws_size;

    const float* x        = (const float*)d_in[0];          // [N,F]
    const int*   eidx     = (const int*)d_in[1];            // [2,E]
    const float* Ws       = (const float*)d_in[2];          // [L,F,H]
    const float* att_src  = (const float*)d_in[3];          // [L,H]
    const float* att_dst  = (const float*)d_in[4];          // [L,H]
    const float* biases   = (const float*)d_in[5];          // [L,H]
    const float* W_out    = (const float*)d_in[6];          // [H,O]
    const float* b_out    = (const float*)d_in[7];          // [O]
    float*       out      = (float*)d_out;                  // [N,O]

    const int* src = eidx;
    const int* dst = eidx + GE;

    // ---- workspace layout (256B aligned slices) ----
    char* ws = (char*)d_ws;
    size_t off = 0;
    auto take = [&](size_t bytes) -> char* {
        char* p = ws + off;
        off = (off + bytes + 255) & ~(size_t)255;
        return p;
    };
    float*    zbuf   = (float*)take(sizeof(float) * GN * GH);
    float*    hcur   = (float*)take(sizeof(float) * GN * GH);
    float*    accum  = (float*)take(sizeof(float) * GN * GH);
    float*    asb    = (float*)take(sizeof(float) * GN);
    float*    adb    = (float*)take(sizeof(float) * GN);
    float*    ssum   = (float*)take(sizeof(float) * GN);
    unsigned* mkey   = (unsigned*)take(sizeof(unsigned) * GN);
    float*    logits = (float*)take(sizeof(float) * (GE + GN));
    _Float16* h16    = (_Float16*)take(sizeof(_Float16) * GN * GF);
    _Float16* w16s   = (_Float16*)take(sizeof(_Float16) * GF * GH);

    const int BLK = 256;
    const int totalE = GE + GN;
    const int Mtiles = GN / 16;                 // 1250
    const int gxM    = (Mtiles + MT - 1) / MT;  // GEMM grid.x

    // layer 0 input: convert x once
    cvt4_f16<<<(GN * GF / 4 + BLK - 1) / BLK, BLK, 0, stream>>>(h16, x, GN * GF / 4);

    for (int l = 0; l < GL; ++l) {
        // swizzle this layer's W into B-fragment layout
        repack_w<<<(GF * GH + BLK - 1) / BLK, BLK, 0, stream>>>(
            w16s, Ws + (size_t)l * GF * GH, GH);

        // z = h @ W  (WMMA, B in registers across 8 M-tiles)
        {
            dim3 grid(gxM, GH / 64);
            gemm_wmma_swz<GH, false><<<grid, 128, 0, stream>>>(
                h16, w16s, zbuf, nullptr, Mtiles);
        }

        // alpha_s / alpha_d
        alphas_kernel<<<(GN * 32 + BLK - 1) / BLK, BLK, 0, stream>>>(
            zbuf, att_src + (size_t)l * GH, att_dst + (size_t)l * GH, asb, adb);

        // clear segment buffers
        fill4_u32<<<(GN / 4 + BLK - 1) / BLK, BLK, 0, stream>>>(mkey, 0u, GN / 4);
        fill4_f32<<<(GN / 4 + BLK - 1) / BLK, BLK, 0, stream>>>(ssum, 0.f, GN / 4);
        fill4_f32<<<(GN * GH / 4 + BLK - 1) / BLK, BLK, 0, stream>>>(
            accum, 0.f, GN * GH / 4);

        // edge passes
        edge_logits_kernel<<<(totalE + BLK - 1) / BLK, BLK, 0, stream>>>(
            src, dst, asb, adb, logits, mkey);
        edge_accum_kernel<<<(totalE * 32 + BLK - 1) / BLK, BLK, 0, stream>>>(
            src, dst, logits, mkey, zbuf, ssum, accum);

        // h = relu(accum / s + b); also produce f16 for next GEMM
        finalize_kernel<<<(GN * GH / 4 + BLK - 1) / BLK, BLK, 0, stream>>>(
            accum, ssum, biases + (size_t)l * GH, hcur, h16);
    }

    // final linear: out = h @ W_out + b_out  (WMMA, Nc = 64)
    repack_w<<<(GH * GO + BLK - 1) / BLK, BLK, 0, stream>>>(w16s, W_out, GO);
    {
        dim3 grid(gxM, GO / 64);
        gemm_wmma_swz<GO, true><<<grid, 128, 0, stream>>>(
            h16, w16s, out, b_out, Mtiles);
    }
}
